// NTXentLoss_1949915152813
// MI455X (gfx1250) — compile-verified, matrix operand order checked by reading, not run on hardware
//
#include <hip/hip_runtime.h>
#include <stdint.h>

// ---------------------------------------------------------------------------
// NT-Xent loss, fused:  normalize -> (Z Z^T / T) -> masked logsumexp -> mean
// B=4096, D=256, N=8192, T=0.5.  Never materializes the 8192x8192 sim matrix.
// Unit-norm rows => |sim| <= 2  => no max-pass needed for logsumexp.
// ---------------------------------------------------------------------------

typedef __bf16 bf16_t;
typedef bf16_t v16bf __attribute__((ext_vector_type(16)));
typedef float  v8f   __attribute__((ext_vector_type(8)));

#define B_ROWS 4096
#define D_DIM  256
#define N_ROWS 8192
#define INV_T  2.0f
#define NTILES (N_ROWS / 16)        // 512 column tiles of 16
#define WAVES_PER_BLOCK 4

union FragU { v16bf v; uint4 q[2]; };

// --------------------------- Kernel 1: normalize ---------------------------
// One wave per row: 256 floats -> sum of squares -> rsqrt -> bf16 (RNE),
// packed uint4 stores into the workspace panel zn[8192][256] (bf16).
__global__ __launch_bounds__(256)
void ntxent_normalize(const float* __restrict__ zi,
                      const float* __restrict__ zj,
                      unsigned short* __restrict__ zn) {
    const int wave = threadIdx.x >> 5;
    const int lane = threadIdx.x & 31;
    const int row  = blockIdx.x * 8 + wave;

    const float* src = (row < B_ROWS) ? (zi + (size_t)row * D_DIM)
                                      : (zj + (size_t)(row - B_ROWS) * D_DIM);
    const float4* p = reinterpret_cast<const float4*>(src) + lane * 2;
    float4 f0 = p[0];
    float4 f1 = p[1];

    float ss = f0.x*f0.x + f0.y*f0.y + f0.z*f0.z + f0.w*f0.w
             + f1.x*f1.x + f1.y*f1.y + f1.z*f1.z + f1.w*f1.w;
#pragma unroll
    for (int m = 16; m >= 1; m >>= 1) ss += __shfl_xor(ss, m, 32);
    const float rn = rsqrtf(ss);

    float v[8] = { f0.x, f0.y, f0.z, f0.w, f1.x, f1.y, f1.z, f1.w };
    uint32_t pk[4];
#pragma unroll
    for (int k = 0; k < 4; ++k) {
        uint32_t u0 = __float_as_uint(v[2*k]   * rn);
        uint32_t u1 = __float_as_uint(v[2*k+1] * rn);
        u0 = (u0 + 0x7fffu + ((u0 >> 16) & 1u)) >> 16;   // RNE to bf16
        u1 = (u1 + 0x7fffu + ((u1 >> 16) & 1u)) >> 16;
        pk[k] = u0 | (u1 << 16);
    }
    uint4 out; out.x = pk[0]; out.y = pk[1]; out.z = pk[2]; out.w = pk[3];
    reinterpret_cast<uint4*>(zn + (size_t)row * D_DIM)[lane] = out;
}

// ------------------ Kernel 2: fused WMMA GEMM + logsumexp ------------------
// One block = 16 rows.  4 waves split the 512 column tiles.  Each wave keeps
// the 16x256 bf16 A panel in registers (8 K-chunks) and runs 8 WMMAs per
// column tile, then streams exp() into per-row accumulators with diagonal
// masking and positive-pair capture.
__global__ __launch_bounds__(128)
void ntxent_main(const unsigned short* __restrict__ zn,
                 float* __restrict__ partials) {
    __shared__ float s_sum[WAVES_PER_BLOCK][16];
    __shared__ float s_pos[WAVES_PER_BLOCK][16];

    const int wave = threadIdx.x >> 5;
    const int lane = threadIdx.x & 31;
    const int half = lane >> 4;      // which 16-lane half of the wave
    const int m    = lane & 15;      // in-tile column / row-slot index

    const int rb     = blockIdx.x * 16;   // row base of this block
    const int pos_cb = rb ^ B_ROWS;       // column tile holding the positives

    // A fragments: row rb+m, K-offset half*8 (+16), for all 8 K-chunks.
    FragU afrag[8];
    {
        const unsigned short* arow = zn + (size_t)(rb + m) * D_DIM + half * 8;
#pragma unroll
        for (int kc = 0; kc < 8; ++kc) {
            const uint4* p = reinterpret_cast<const uint4*>(arow + kc * 32);
            afrag[kc].q[0] = p[0];    // K +0..7   -> frag elems 0..7
            afrag[kc].q[1] = p[2];    // K +16..23 -> frag elems 8..15
        }
    }

    float sumacc[8], posacc[8];
#pragma unroll
    for (int v = 0; v < 8; ++v) { sumacc[v] = 0.0f; posacc[v] = 0.0f; }

    for (int ct = wave; ct < NTILES; ct += WAVES_PER_BLOCK) {
        const int cb = ct * 16;
        // B operand (column-major tile of Z Z^T) == same load pattern as A.
        const unsigned short* brow = zn + (size_t)(cb + m) * D_DIM + half * 8;
        v8f c = {};
#pragma unroll
        for (int kc = 0; kc < 8; ++kc) {
            FragU b;
            const uint4* p = reinterpret_cast<const uint4*>(brow + kc * 32);
            b.q[0] = p[0];
            b.q[1] = p[2];
            c = __builtin_amdgcn_wmma_f32_16x16x32_bf16(
                    false, afrag[kc].v, false, b.v, (short)0, c, false, false);
        }
        const bool diagTile = (cb == rb);
        const bool posTile  = (cb == pos_cb);
#pragma unroll
        for (int v = 0; v < 8; ++v) {
            // C layout: VGPR v holds (M = v + 8*half, N = m)
            const bool onDiagLane = (m == (v + half * 8));
            const float x = c[v] * INV_T;
            const float e = (diagTile && onDiagLane) ? 0.0f : __expf(x);
            sumacc[v] += e;
            if (posTile && onDiagLane) posacc[v] += x;
        }
    }

    // Reduce over the 16 lanes of each half (masks <16 stay inside the half).
#pragma unroll
    for (int v = 0; v < 8; ++v) {
        float s = sumacc[v], q = posacc[v];
#pragma unroll
        for (int msk = 8; msk >= 1; msk >>= 1) {
            s += __shfl_xor(s, msk, 32);
            q += __shfl_xor(q, msk, 32);
        }
        if (m == 0) {
            s_sum[wave][half * 8 + v] = s;
            s_pos[wave][half * 8 + v] = q;
        }
    }
    __syncthreads();

    // Combine the 4 waves, compute per-row loss, reduce, write block partial.
    if (threadIdx.x < 16) {
        const int j = threadIdx.x;
        float tot = 0.0f, pos = 0.0f;
#pragma unroll
        for (int w = 0; w < WAVES_PER_BLOCK; ++w) {
            tot += s_sum[w][j];
            pos += s_pos[w][j];
        }
        float loss = __logf(tot) - pos;
#pragma unroll
        for (int msk = 8; msk >= 1; msk >>= 1) loss += __shfl_xor(loss, msk, 32);
        if (j == 0) partials[blockIdx.x] = loss * (1.0f / (float)N_ROWS);
    }
}

// ------------------------- Kernel 3: final reduce --------------------------
// Fixed-order tree reduction of the 512 block partials (deterministic).
__global__ __launch_bounds__(512)
void ntxent_finalize(const float* __restrict__ partials,
                     float* __restrict__ out) {
    __shared__ float s[512];
    s[threadIdx.x] = partials[threadIdx.x];
    __syncthreads();
    for (int stride = 256; stride >= 1; stride >>= 1) {
        if (threadIdx.x < stride) s[threadIdx.x] += s[threadIdx.x + stride];
        __syncthreads();
    }
    if (threadIdx.x == 0) out[0] = s[0];
}

// ---------------------------------------------------------------------------
extern "C" void kernel_launch(void* const* d_in, const int* in_sizes, int n_in,
                              void* d_out, int out_size, void* d_ws, size_t ws_size,
                              hipStream_t stream) {
    (void)in_sizes; (void)n_in; (void)out_size; (void)ws_size;
    const float* zi = (const float*)d_in[0];
    const float* zj = (const float*)d_in[1];

    unsigned short* zn = (unsigned short*)d_ws;                    // 4 MiB bf16 panel
    float* partials = (float*)((char*)d_ws + (size_t)N_ROWS * D_DIM * 2);  // 512 f32
    float* out = (float*)d_out;

    ntxent_normalize<<<N_ROWS / 8, 256, 0, stream>>>(zi, zj, zn);
    ntxent_main<<<NTILES, 128, 0, stream>>>(zn, partials);
    ntxent_finalize<<<1, 512, 0, stream>>>(partials, out);
}
